// ChebKernelMixture_60103772340308
// MI455X (gfx1250) — matrix operand order repaired and checked by reading, not compiled, not run on hardware
//
#include <hip/hip_runtime.h>

typedef __attribute__((ext_vector_type(8)))  _Float16 v8h;
typedef __attribute__((ext_vector_type(16))) _Float16 v16h;
typedef __attribute__((ext_vector_type(8)))  float    v8f;

#define P_PTS 16384
#define KF 64              // features 1..64 (constant feature 0 folded into w0)
#define BLK 128            // output tile edge per workgroup
#define LDS_STRIDE 72      // halfs per LDS row (bank-conflict-free, 16B aligned rows)

__device__ __forceinline__ v16h cat8(v8h a, v8h b) {
    return __builtin_shufflevector(a, b, 0, 1, 2, 3, 4, 5, 6, 7,
                                         8, 9, 10, 11, 12, 13, 14, 15);
}

// ---------------- Phase 1: build split-fp16 psi (P x 64 hi, P x 64 lo) + w0 ----------------
__global__ __launch_bounds__(256) void psi_kernel(const float* __restrict__ xs,
                                                  const float* __restrict__ logits,
                                                  _Float16* __restrict__ psi_h,
                                                  _Float16* __restrict__ psi_l,
                                                  float* __restrict__ w0_out) {
    __shared__ float sw[33];
    if (threadIdx.x == 0) {
        float mx = logits[0];
        for (int n = 1; n < 33; ++n) mx = fmaxf(mx, logits[n]);
        float sum = 0.f;
        for (int n = 0; n < 33; ++n) { float e = __expf(logits[n] - mx); sw[n] = e; sum += e; }
        float inv = 1.f / sum;
        if (blockIdx.x == 0) *w0_out = sw[0] * inv;          // w0 = sqrt(w0)^2
        for (int n = 0; n < 33; ++n) sw[n] = sqrtf(sw[n] * inv);
    }
    __syncthreads();

    int i = blockIdx.x * blockDim.x + threadIdx.x;
    float x = xs[i];
    x = fminf(1.f, fmaxf(-1.f, x));
    float s = sqrtf(fmaxf(0.f, 1.f - x * x));

    _Float16* rh = psi_h + (size_t)i * KF;
    _Float16* rl = psi_l + (size_t)i * KF;

    float Tprev = 1.f, T = x;        // T_0, T_1
    float Uprev = 1.f, U = 2.f * x;  // U_0, U_1
#pragma unroll
    for (int n = 1; n <= 32; ++n) {
        float f0 = sw[n] * T;            // feature 2n-1 -> slot 2n-2
        float f1 = sw[n] * s * Uprev;    // feature 2n   -> slot 2n-1
        _Float16 h0 = (_Float16)f0;  _Float16 l0 = (_Float16)(f0 - (float)h0);
        _Float16 h1 = (_Float16)f1;  _Float16 l1 = (_Float16)(f1 - (float)h1);
        rh[2 * n - 2] = h0;  rl[2 * n - 2] = l0;
        rh[2 * n - 1] = h1;  rl[2 * n - 1] = l1;
        float Tn = 2.f * x * T - Tprev; Tprev = T; T = Tn;
        float Un = 2.f * x * U - Uprev; Uprev = U; U = Un;
    }
}

// ---------------- Phase 2: gram = w0 + psi @ psi^T via split-fp16 WMMA ----------------
__global__ __launch_bounds__(256) void gram_kernel(const _Float16* __restrict__ psi_h,
                                                   const _Float16* __restrict__ psi_l,
                                                   const float* __restrict__ w0p,
                                                   float* __restrict__ out) {
    __shared__ __align__(16) _Float16 BsH[BLK * LDS_STRIDE];
    __shared__ __align__(16) _Float16 BsL[BLK * LDS_STRIDE];

    const int tid  = threadIdx.x;
    const int lane = tid & 31;
    const int wave = tid >> 5;        // 0..7 : row sub-tile of this wave
    const int half = lane >> 4;       // lane half selects K sub-ranges
    const int mrow = lane & 15;       // M (A rows) / N (B,C columns)
    const int i0 = blockIdx.y * BLK;
    const int j0 = blockIdx.x * BLK;
    const float w0 = *w0p;            // uniform scalar

    // Cooperative staging of B tile: 128 rows x 64 halfs (8 x 16B chunks/row), hi + lo
    for (int t = tid; t < BLK * 8; t += 256) {
        int r = t >> 3, c = t & 7;
        uint4 vh = *(const uint4*)(psi_h + (size_t)(j0 + r) * KF + c * 8);
        uint4 vl = *(const uint4*)(psi_l + (size_t)(j0 + r) * KF + c * 8);
        *(uint4*)(&BsH[r * LDS_STRIDE + c * 8]) = vh;
        *(uint4*)(&BsL[r * LDS_STRIDE + c * 8]) = vl;
    }

    // Preload this wave's A strips (hi & lo) into registers.
    // f16 A 16x32 layout: lane holds K = 8*half+{0..7} (VGPR0-3) and 16+8*half+{0..7} (VGPR4-7).
    v16h hA[2], lA[2];
    {
        const _Float16* rh = psi_h + (size_t)(i0 + wave * 16 + mrow) * KF;
        const _Float16* rl = psi_l + (size_t)(i0 + wave * 16 + mrow) * KF;
#pragma unroll
        for (int c = 0; c < 2; ++c) {
            v8h p0 = *(const v8h*)(rh + 32 * c + 8 * half);
            v8h p1 = *(const v8h*)(rh + 32 * c + 16 + 8 * half);
            hA[c] = cat8(p0, p1);
            v8h q0 = *(const v8h*)(rl + 32 * c + 8 * half);
            v8h q1 = *(const v8h*)(rl + 32 * c + 16 + 8 * half);
            lA[c] = cat8(q0, q1);
        }
    }

    v8f acc[8];
#pragma unroll
    for (int j = 0; j < 8; ++j)
#pragma unroll
        for (int k = 0; k < 8; ++k) acc[j][k] = w0;   // fold constant feature 0

    __syncthreads();

#pragma unroll
    for (int c = 0; c < 2; ++c) {
#pragma unroll
        for (int j = 0; j < 8; ++j) {
            // f16 B 32x16 layout: lane = column N, K = 8*half+{0..7} and 16+8*half+{0..7}.
            const _Float16* bh = &BsH[(j * 16 + mrow) * LDS_STRIDE + 32 * c + 8 * half];
            const _Float16* bl = &BsL[(j * 16 + mrow) * LDS_STRIDE + 32 * c + 8 * half];
            v16h bH = cat8(*(const v8h*)bh, *(const v8h*)(bh + 16));
            v16h bL = cat8(*(const v8h*)bl, *(const v8h*)(bl + 16));

            acc[j] = __builtin_amdgcn_wmma_f32_16x16x32_f16(
                false, hA[c], false, bH, (short)0, acc[j], false, false);
            acc[j] = __builtin_amdgcn_wmma_f32_16x16x32_f16(
                false, hA[c], false, bL, (short)0, acc[j], false, false);
            acc[j] = __builtin_amdgcn_wmma_f32_16x16x32_f16(
                false, lA[c], false, bH, (short)0, acc[j], false, false);
        }
    }

    // C/D 16x16 f32 layout: VGPR v -> M = v + 8*half; N = lane&15.
#pragma unroll
    for (int j = 0; j < 8; ++j) {
#pragma unroll
        for (int v = 0; v < 8; ++v) {
            const int r = i0 + wave * 16 + v + 8 * half;
            const int col = j0 + j * 16 + mrow;
            out[(size_t)r * P_PTS + col] = acc[j][v];
        }
    }
}

extern "C" void kernel_launch(void* const* d_in, const int* in_sizes, int n_in,
                              void* d_out, int out_size, void* d_ws, size_t ws_size,
                              hipStream_t stream) {
    const float* xs     = (const float*)d_in[0];
    const float* logits = (const float*)d_in[1];
    float* out = (float*)d_out;

    _Float16* psi_h = (_Float16*)d_ws;                       // 2 MB
    _Float16* psi_l = psi_h + (size_t)P_PTS * KF;            // 2 MB
    float*    w0p   = (float*)(psi_l + (size_t)P_PTS * KF);  // 4 B

    psi_kernel<<<P_PTS / 256, 256, 0, stream>>>(xs, logits, psi_h, psi_l, w0p);

    dim3 grid(P_PTS / BLK, P_PTS / BLK);
    gram_kernel<<<grid, 256, 0, stream>>>(psi_h, psi_l, w0p, out);
}
